// Matcher_57045755625774
// MI455X (gfx1250) — compile-verified, matrix operand order checked by reading, not run on hardware
//
#include <hip/hip_runtime.h>
#include <float.h>

typedef __attribute__((ext_vector_type(16))) _Float16 v16h;
typedef __attribute__((ext_vector_type(8)))  float    v8f;

#define CD 64
#define HWD 4096
#define NCH 10
#define QCHUNKS (HWD / 16)        // 256
#define FRAGSTRIDE (4 * 2 * 32)   // v16h slots per q-chunk in packed layout

// Packed A layout in d_ws (half elements):
//   idx = ((((b*QCHUNKS + qc)*4 + t)*2 + f)*32 + lane)*16 + i
// Fragment element i<8  <-> K = f*32 + hi*8 + i
//          element i>=8 <-> K = f*32 + 16 + hi*8 + (i-8)
// where hi = lane>>4, row M = lane&15 (ISA 16-bit A 16x32 layout).

// ---------------------------------------------------------------------------
// Pack kernel: f32 temps (B,HW,C) -> f16 WMMA A fragments in d_ws. One-time.
// ---------------------------------------------------------------------------
__global__ __launch_bounds__(256) void pack_kernel(
    const float* __restrict__ t0, const float* __restrict__ t1,
    const float* __restrict__ t2, const float* __restrict__ t3,
    _Float16* __restrict__ ws)
{
    const int tid  = blockIdx.x * blockDim.x + threadIdx.x;
    const int lane = tid & 31;
    const int f    = (tid >> 5) & 1;
    const int t    = (tid >> 6) & 3;
    const int qc   = (tid >> 8) & (QCHUNKS - 1);
    const int b    = tid >> 16;

    const float* temps[4] = { t0, t1, t2, t3 };
    const int hi = lane >> 4;
    const int m  = lane & 15;

    const float* src = temps[t] + ((size_t)(b * HWD + qc * 16 + m)) * CD
                                + f * 32 + hi * 8;
    const float4 p0 = ((const float4*)src)[0];
    const float4 p1 = ((const float4*)src)[1];
    const float4 p2 = ((const float4*)(src + 16))[0];
    const float4 p3 = ((const float4*)(src + 16))[1];

    v16h a;
    a[0]=(_Float16)p0.x;  a[1]=(_Float16)p0.y;  a[2]=(_Float16)p0.z;  a[3]=(_Float16)p0.w;
    a[4]=(_Float16)p1.x;  a[5]=(_Float16)p1.y;  a[6]=(_Float16)p1.z;  a[7]=(_Float16)p1.w;
    a[8]=(_Float16)p2.x;  a[9]=(_Float16)p2.y;  a[10]=(_Float16)p2.z; a[11]=(_Float16)p2.w;
    a[12]=(_Float16)p3.x; a[13]=(_Float16)p3.y; a[14]=(_Float16)p3.z; a[15]=(_Float16)p3.w;

    *((v16h*)ws + (size_t)tid) = a;
}

// ---------------------------------------------------------------------------
// Fine kernel helpers
// ---------------------------------------------------------------------------
__device__ __forceinline__ void loadA(const v16h* pAc, v16h (&fa)[8]) {
    #pragma unroll
    for (int j = 0; j < 8; ++j) fa[j] = pAc[j * 32];   // 2x b128 each
}

__device__ __forceinline__ void loadW(const float* sdsb, int q0, int hi,
                                      float (&w)[8]) {
    #pragma unroll
    for (int m = 0; m < 8; ++m)
        w[m] = sdsb[(size_t)(q0 + hi * 8 + m) * HWD];
}

__device__ __forceinline__ void computeChunk(
    const v16h (&fa)[8], const v16h& fb0, const v16h& fb1,
    const float (&w)[8], float (&runmax)[4][8])
{
    #pragma unroll
    for (int t = 0; t < 4; ++t) {
        v8f c = {};
        c = __builtin_amdgcn_wmma_f32_16x16x32_f16(
                false, fa[t * 2 + 0], false, fb0, (short)0, c, false, false);
        c = __builtin_amdgcn_wmma_f32_16x16x32_f16(
                false, fa[t * 2 + 1], false, fb1, (short)0, c, false, false);
        #pragma unroll
        for (int m = 0; m < 8; ++m) {
            float cv = c[m];
            if (t >= 2) cv *= w[m];       // local scores: sds weighting
            // raw v_max_num_f32: skip fmaxf's NaN-canonicalization self-max
            asm("v_max_num_f32 %0, %1, %2"
                : "=v"(runmax[t][m]) : "v"(runmax[t][m]), "v"(cv));
        }
    }
}

// ---------------------------------------------------------------------------
// Fine kernel: fused GEMM (f16 WMMA, f32 accum) + sds weighting + column max.
// Grid: (B=4, ktiles=32). Block: 256 = 8 waves; wave owns 16 k-columns.
// Double-buffered A/w registers: chunk n+1 loads issue before chunk n WMMAs.
// ---------------------------------------------------------------------------
__global__ __launch_bounds__(256) void fine_kernel(
    const float* __restrict__ key,        // (B, C, HW) f32
    const float* __restrict__ sds,        // (B, HW, HW) f32
    const _Float16* __restrict__ wsA,     // packed A fragments
    float* __restrict__ out)              // (B, 10, HW)
{
    const int b     = blockIdx.x;
    const int ktile = blockIdx.y;
    const int wave  = threadIdx.x >> 5;
    const int lane  = threadIdx.x & 31;
    const int n     = lane & 15;
    const int hi    = lane >> 4;

    const int kcol = ktile * 128 + wave * 16 + n;

    // ---- B fragments from key (one-time f32->f16, key is L2-resident) ----
    v16h fb0, fb1;
    {
        const float* kb = key + (size_t)b * CD * HWD + kcol;
        #pragma unroll
        for (int i = 0; i < 8; ++i) {
            const int c0 = hi * 8 + i;
            fb0[i]     = (_Float16)kb[(size_t)(c0     ) * HWD];
            fb0[i + 8] = (_Float16)kb[(size_t)(c0 + 16) * HWD];
            fb1[i]     = (_Float16)kb[(size_t)(c0 + 32) * HWD];
            fb1[i + 8] = (_Float16)kb[(size_t)(c0 + 48) * HWD];
        }
    }

    float runmax[4][8];
    #pragma unroll
    for (int t = 0; t < 4; ++t)
        #pragma unroll
        for (int m = 0; m < 8; ++m)
            runmax[t][m] = -FLT_MAX;

    const float* sdsb = sds + (size_t)b * HWD * HWD + kcol;
    const v16h*  pA   = (const v16h*)wsA + ((size_t)b * QCHUNKS) * FRAGSTRIDE + lane;

    v16h fa0[8], fa1[8];
    float w0[8], w1[8];

    // prologue: chunk 0
    loadA(pA, fa0);
    loadW(sdsb, 0, hi, w0);

    // steady state: issue next chunk's loads, then compute current chunk
    for (int qc = 0; qc < QCHUNKS - 2; qc += 2) {
        loadA(pA + (size_t)(qc + 1) * FRAGSTRIDE, fa1);
        loadW(sdsb, (qc + 1) * 16, hi, w1);
        computeChunk(fa0, fb0, fb1, w0, runmax);

        loadA(pA + (size_t)(qc + 2) * FRAGSTRIDE, fa0);
        loadW(sdsb, (qc + 2) * 16, hi, w0);
        computeChunk(fa1, fb0, fb1, w1, runmax);
    }
    // epilogue: chunks QCHUNKS-2 (in fa0/w0) and QCHUNKS-1
    loadA(pA + (size_t)(QCHUNKS - 1) * FRAGSTRIDE, fa1);
    loadW(sdsb, (QCHUNKS - 1) * 16, hi, w1);
    computeChunk(fa0, fb0, fb1, w0, runmax);
    computeChunk(fa1, fb0, fb1, w1, runmax);

    // ---- final column-max reduction + store (each element written once) ---
    #pragma unroll
    for (int t = 0; t < 4; ++t) {
        float r = runmax[t][0];
        #pragma unroll
        for (int m = 1; m < 8; ++m)
            asm("v_max_num_f32 %0, %1, %2" : "=v"(r) : "v"(r), "v"(runmax[t][m]));
        const float o = __shfl_xor(r, 16, 32);   // combine M=0..7 with M=8..15
        asm("v_max_num_f32 %0, %1, %2" : "=v"(r) : "v"(r), "v"(o));
        if (hi == 0)
            out[((size_t)b * NCH + t) * HWD + kcol] = r;
    }
}

// ---------------------------------------------------------------------------
// Coarse kernel: six (1,64)x(64,4096) matvecs per batch. Trivially BW-bound.
// ---------------------------------------------------------------------------
__global__ __launch_bounds__(256) void coarse_kernel(
    const float* __restrict__ key,
    const float* __restrict__ t0, const float* __restrict__ t1,
    const float* __restrict__ t2, const float* __restrict__ t3,
    const float* __restrict__ t4, const float* __restrict__ t5,
    float* __restrict__ out)
{
    const int idx = blockIdx.x * blockDim.x + threadIdx.x;  // 0 .. B*HW-1
    const int b   = idx >> 12;
    const int pos = idx & (HWD - 1);

    const float* kb = key + (size_t)b * CD * HWD + pos;
    const float* tp[6] = { t0 + b * CD, t1 + b * CD, t2 + b * CD,
                           t3 + b * CD, t4 + b * CD, t5 + b * CD };
    float acc[6] = {0.f, 0.f, 0.f, 0.f, 0.f, 0.f};

    for (int c = 0; c < CD; ++c) {
        const float kv = kb[(size_t)c * HWD];
        #pragma unroll
        for (int j = 0; j < 6; ++j) acc[j] += tp[j][c] * kv;
    }
    #pragma unroll
    for (int j = 0; j < 6; ++j)
        out[((size_t)b * NCH + 4 + j) * HWD + pos] = acc[j];
}

extern "C" void kernel_launch(void* const* d_in, const int* in_sizes, int n_in,
                              void* d_out, int out_size, void* d_ws, size_t ws_size,
                              hipStream_t stream) {
    const float* key  = (const float*)d_in[0];
    const float* sds  = (const float*)d_in[1];
    const float* gbg  = (const float*)d_in[2];
    const float* gfg  = (const float*)d_in[3];
    const float* lbg  = (const float*)d_in[4];
    const float* lfg  = (const float*)d_in[5];
    const float* obg  = (const float*)d_in[6];
    const float* ofg  = (const float*)d_in[7];
    const float* sbg  = (const float*)d_in[8];
    const float* sfg  = (const float*)d_in[9];
    const float* lobg = (const float*)d_in[10];
    const float* lofg = (const float*)d_in[11];
    float* out = (float*)d_out;
    _Float16* wsA = (_Float16*)d_ws;   // 8 MB of packed f16 A fragments

    const int pack_threads = 4 * QCHUNKS * 4 * 2 * 32;
    pack_kernel<<<pack_threads / 256, 256, 0, stream>>>(gbg, gfg, lbg, lfg, wsA);

    dim3 fgrid(4, 32);
    fine_kernel<<<fgrid, 256, 0, stream>>>(key, sds, wsA, out);

    coarse_kernel<<<(4 * HWD) / 256, 256, 0, stream>>>(
        key, obg, ofg, sbg, sfg, lobg, lofg, out);
}
